// QuantizedLoRALinear_31095563223128
// MI455X (gfx1250) — matrix operand order repaired and checked by reading, not compiled, not run on hardware
//
#include <hip/hip_runtime.h>
#include <stdint.h>

typedef __attribute__((ext_vector_type(16))) __bf16 v16bf;
typedef __attribute__((ext_vector_type(8)))  float  v8f;
typedef __attribute__((ext_vector_type(4)))  unsigned int v4u;
typedef __attribute__((ext_vector_type(8)))  int v8i;
typedef __attribute__((ext_vector_type(4)))  int v4i;

namespace {
constexpr int   M_TOT  = 4 * 2048;        // B * S
constexpr int   K_TOT  = 4096;            // D_IN
constexpr int   N_TOT  = 4096;            // D_OUT
constexpr int   R_LORA = 64;
constexpr int   K_PACK = K_TOT + R_LORA;  // 4160: base K + fused LoRA K
constexpr float LORA_SCALING = 16.0f / 64.0f;

constexpr int BM = 128;
constexpr int BN = 128;
constexpr int BK = 64;
constexpr int KP = 72;   // padded LDS pitch (bf16 elems): 144 B -> conflict-free b128
}

__constant__ float NF4_TAB[16] = {
    -1.0f, -0.6962f, -0.5251f, -0.3949f, -0.2844f, -0.1848f,
    -0.0911f, 0.0f, 0.0796f, 0.1609f, 0.2461f, 0.3379f,
    0.4407f, 0.5626f, 0.723f, 1.0f};

union Frag { v16bf v; uint4 q[2]; };

__device__ __forceinline__ uint32_t pack2_bf16(float a, float b) {
  return (__float_as_uint(a) >> 16) | (__float_as_uint(b) & 0xffff0000u);
}
__device__ __forceinline__ unsigned short bf16_of(float a) {
  return (unsigned short)(__float_as_uint(a) >> 16);
}
__device__ __forceinline__ uint4 pack8_bf16(float4 f0, float4 f1) {
  uint4 q;
  q.x = pack2_bf16(f0.x, f0.y);
  q.y = pack2_bf16(f0.z, f0.w);
  q.z = pack2_bf16(f1.x, f1.y);
  q.w = pack2_bf16(f1.z, f1.w);
  return q;
}
__device__ __forceinline__ uint32_t lds_addr_of(const void* p) {
  // generic LDS address: low 32 bits are the LDS byte offset (flat aperture rule)
  return (uint32_t)(uintptr_t)p;
}

// One 16x16x32 k-step over a wave's 64x32 sub-tile (4x2 f32 accumulators).
__device__ __forceinline__ void wmma_step(const unsigned short* __restrict__ sA,
                                          const unsigned short* __restrict__ sB,
                                          int wm, int wn, int lane, int kk,
                                          v8f (&acc)[4][2]) {
  Frag a[4], b[2];
  const int l15 = lane & 15;
  const int akb = kk + ((lane >> 4) << 3);
#pragma unroll
  for (int mi = 0; mi < 4; ++mi) {
    const unsigned short* p = sA + (wm * 64 + mi * 16 + l15) * KP + akb;
    a[mi].q[0] = *(const uint4*)(p);
    a[mi].q[1] = *(const uint4*)(p + 16);
  }
  const int bkb = kk + ((lane >> 4) << 3);
#pragma unroll
  for (int ni = 0; ni < 2; ++ni) {
    const unsigned short* p = sB + (wn * 32 + ni * 16 + l15) * KP + bkb;
    b[ni].q[0] = *(const uint4*)(p);
    b[ni].q[1] = *(const uint4*)(p + 16);
  }
#pragma unroll
  for (int mi = 0; mi < 4; ++mi)
#pragma unroll
    for (int ni = 0; ni < 2; ++ni)
      acc[mi][ni] = __builtin_amdgcn_wmma_f32_16x16x32_bf16(
          false, a[mi].v, false, b[ni].v, (short)0, acc[mi][ni], false, false);
}

// ---------------------------------------------------------------------------
// TDM: descriptor-driven 2D tile load (128 x 64 bf16) global -> LDS with
// hardware padding (4 DWORDs after every 32 DWORDs) == KP=72 pitch.
// ---------------------------------------------------------------------------
#if __has_builtin(__builtin_amdgcn_tensor_load_to_lds)
#define HAVE_TDM 1
__device__ __forceinline__ void tdm_load_tile(const unsigned short* gbase,
                                              uint32_t lds_byte, int rows) {
  const uint64_t ga = (uint64_t)(uintptr_t)gbase;
  v4u g0 = {1u, lds_byte, (uint32_t)ga,
            (uint32_t)((ga >> 32) & 0x01FFFFFFu) | (2u << 30)};
  v8i g1 = {(int)((1u << 16) | (1u << 20) | (4u << 22) | (3u << 25)),
            (int)((uint32_t)K_PACK << 16),
            (int)((uint32_t)rows << 16),
            (int)(64u << 16),
            (int)128u,
            (int)K_PACK,
            0, 0};
  v4i z4 = {0, 0, 0, 0};
#if defined(__clang_major__) && (__clang_major__ >= 23)
  v8i z8 = {0, 0, 0, 0, 0, 0, 0, 0};
  __builtin_amdgcn_tensor_load_to_lds(g0, g1, z4, z4, z8, 0);
#else
  __builtin_amdgcn_tensor_load_to_lds(g0, g1, z4, z4, 0);
#endif
}
#else
#define HAVE_TDM 0
#endif

// ---------------------------------------------------------------------------
// Pre-pack kernels (one-shot, HBM-rate).
// ---------------------------------------------------------------------------
__global__ __launch_bounds__(256)
void pack_x_kernel(const float* __restrict__ x, unsigned short* __restrict__ xb) {
  const long g   = (long)blockIdx.x * blockDim.x + threadIdx.x;  // 8 elems each
  const long row = g >> 9;                                       // 512 chunks/row
  const int  c   = (int)(g & 511);
  const float4 f0 = *(const float4*)(x + row * K_TOT + c * 8);
  const float4 f1 = *(const float4*)(x + row * K_TOT + c * 8 + 4);
  *(uint4*)(xb + row * K_PACK + c * 8) = pack8_bf16(f0, f1);
}

__global__ __launch_bounds__(256)
void pack_w_kernel(const int* __restrict__ qw, const float* __restrict__ wscale,
                   unsigned short* __restrict__ wb) {
  __shared__ unsigned short sLUT[16];
  if (threadIdx.x < 16) sLUT[threadIdx.x] = bf16_of(NF4_TAB[threadIdx.x] * wscale[0]);
  __syncthreads();
  const long g   = (long)blockIdx.x * blockDim.x + threadIdx.x;
  const long row = g >> 9;
  const int  c   = (int)(g & 511);
  const int4 q0 = *(const int4*)(qw + row * K_TOT + c * 8);
  const int4 q1 = *(const int4*)(qw + row * K_TOT + c * 8 + 4);
  uint4 p;
  p.x = (uint32_t)sLUT[q0.x & 15] | ((uint32_t)sLUT[q0.y & 15] << 16);
  p.y = (uint32_t)sLUT[q0.z & 15] | ((uint32_t)sLUT[q0.w & 15] << 16);
  p.z = (uint32_t)sLUT[q1.x & 15] | ((uint32_t)sLUT[q1.y & 15] << 16);
  p.w = (uint32_t)sLUT[q1.z & 15] | ((uint32_t)sLUT[q1.w & 15] << 16);
  *(uint4*)(wb + row * K_PACK + c * 8) = p;
}

__global__ __launch_bounds__(256)
void pack_wtail_kernel(const float* __restrict__ loraB, unsigned short* __restrict__ wb) {
  const long g   = (long)blockIdx.x * blockDim.x + threadIdx.x;  // 8 elems each
  const long row = g >> 3;                                       // 8 chunks/row
  const int  c   = (int)(g & 7);
  const float4 f0 = *(const float4*)(loraB + row * R_LORA + c * 8);
  const float4 f1 = *(const float4*)(loraB + row * R_LORA + c * 8 + 4);
  *(uint4*)(wb + row * K_PACK + K_TOT + c * 8) = pack8_bf16(f0, f1);
}

__global__ __launch_bounds__(256)
void pack_a_kernel(const float* __restrict__ loraA, unsigned short* __restrict__ ab) {
  const long g   = (long)blockIdx.x * blockDim.x + threadIdx.x;  // 8 elems each
  const long off = g * 8;                                        // flat 64*4096
  const float4 f0 = *(const float4*)(loraA + off);
  const float4 f1 = *(const float4*)(loraA + off + 4);
  *(uint4*)(ab + off) = pack8_bf16(f0, f1);
}

// ---------------------------------------------------------------------------
// LoRA stage 1, fast path: all-bf16 operands (xb packed, ab packed).
// t[m, r] = LORA_SCALING * sum_k xb[m,k] * ab[r,k], written into xb tail cols.
// ---------------------------------------------------------------------------
__global__ __launch_bounds__(256, 4)
void lora_xa_bf16_kernel(const unsigned short* __restrict__ xb,
                         const unsigned short* __restrict__ ab,
                         unsigned short* __restrict__ dst /* = xb + K_TOT */) {
  const int tid  = threadIdx.x;
  const int lane = tid & 31;
  const int wave = tid >> 5;
  const long m0  = (long)blockIdx.x * 128 + wave * 16;

  v8f acc[4];
#pragma unroll
  for (int i = 0; i < 4; ++i) acc[i] = (v8f){0, 0, 0, 0, 0, 0, 0, 0};

  const int  l15  = lane & 15;
  const long arow = m0 + l15;

  for (int k = 0; k < K_TOT; k += 32) {
    Frag a;
    const int akb = k + ((lane >> 4) << 3);
    a.q[0] = *(const uint4*)(xb + arow * K_PACK + akb);
    a.q[1] = *(const uint4*)(xb + arow * K_PACK + akb + 16);

    const int bkb = k + ((lane >> 4) << 3);
#pragma unroll
    for (int ni = 0; ni < 4; ++ni) {
      Frag b;
      const long r = ni * 16 + l15;
      b.q[0] = *(const uint4*)(ab + r * K_TOT + bkb);
      b.q[1] = *(const uint4*)(ab + r * K_TOT + bkb + 16);
      acc[ni] = __builtin_amdgcn_wmma_f32_16x16x32_bf16(
          false, a.v, false, b.v, (short)0, acc[ni], false, false);
    }
  }
  const int rofs = (lane >> 4) * 8;
#pragma unroll
  for (int ni = 0; ni < 4; ++ni)
#pragma unroll
    for (int v = 0; v < 8; ++v)
      dst[(m0 + v + rofs) * (long)K_PACK + ni * 16 + l15] =
          bf16_of(acc[ni][v] * LORA_SCALING);
}

// ---------------------------------------------------------------------------
// LoRA stage 1, fallback: fp32 operands converted on the fly.
// ---------------------------------------------------------------------------
__global__ __launch_bounds__(256, 4)
void lora_xa_f32_kernel(const float* __restrict__ x, const float* __restrict__ loraA,
                        unsigned short* __restrict__ dst, long dstStride) {
  const int tid  = threadIdx.x;
  const int lane = tid & 31;
  const int wave = tid >> 5;
  const long m0  = (long)blockIdx.x * 128 + wave * 16;

  v8f acc[4];
#pragma unroll
  for (int i = 0; i < 4; ++i) acc[i] = (v8f){0, 0, 0, 0, 0, 0, 0, 0};

  const int  l15  = lane & 15;
  const long arow = m0 + l15;

  for (int k = 0; k < K_TOT; k += 32) {
    Frag a;
    const int akb = k + ((lane >> 4) << 3);
    const float4 f0 = *(const float4*)(x + arow * K_TOT + akb);
    const float4 f1 = *(const float4*)(x + arow * K_TOT + akb + 4);
    const float4 f2 = *(const float4*)(x + arow * K_TOT + akb + 16);
    const float4 f3 = *(const float4*)(x + arow * K_TOT + akb + 20);
    a.q[0] = pack8_bf16(f0, f1);
    a.q[1] = pack8_bf16(f2, f3);

    const int bkb = k + ((lane >> 4) << 3);
#pragma unroll
    for (int ni = 0; ni < 4; ++ni) {
      Frag b;
      const long r = ni * 16 + l15;
      const float4 g0 = *(const float4*)(loraA + r * K_TOT + bkb);
      const float4 g1 = *(const float4*)(loraA + r * K_TOT + bkb + 4);
      const float4 g2 = *(const float4*)(loraA + r * K_TOT + bkb + 16);
      const float4 g3 = *(const float4*)(loraA + r * K_TOT + bkb + 20);
      b.q[0] = pack8_bf16(g0, g1);
      b.q[1] = pack8_bf16(g2, g3);
      acc[ni] = __builtin_amdgcn_wmma_f32_16x16x32_bf16(
          false, a.v, false, b.v, (short)0, acc[ni], false, false);
    }
  }
  const int rofs = (lane >> 4) * 8;
#pragma unroll
  for (int ni = 0; ni < 4; ++ni)
#pragma unroll
    for (int v = 0; v < 8; ++v)
      dst[(m0 + v + rofs) * dstStride + ni * 16 + l15] =
          bf16_of(acc[ni][v] * LORA_SCALING);
}

#if HAVE_TDM
// ---------------------------------------------------------------------------
// Fast path: pure bf16 GEMM, M=8192 N=4096 K=4160 (LoRA fused into K).
// TDM double-buffered tile staging; hot loop = TDM issue + ds b128 + WMMA.
// ---------------------------------------------------------------------------
__global__ __launch_bounds__(256, 2)
void gemm_bf16_tdm_kernel(const unsigned short* __restrict__ xb,
                          const unsigned short* __restrict__ wb,
                          float* __restrict__ out) {
  __shared__ __align__(16) unsigned short sA[2][BM * KP];
  __shared__ __align__(16) unsigned short sB[2][BN * KP];

  const int tid  = threadIdx.x;
  const int lane = tid & 31;
  const int wave = tid >> 5;
  const int wm   = wave & 1;
  const int wn   = wave >> 1;
  const long m0  = (long)blockIdx.x * BM;
  const long n0  = (long)blockIdx.y * BN;

  v8f acc[4][2];
#pragma unroll
  for (int i = 0; i < 4; ++i)
#pragma unroll
    for (int j = 0; j < 2; ++j) acc[i][j] = (v8f){0, 0, 0, 0, 0, 0, 0, 0};

  constexpr int nKT = K_PACK / BK;  // 65
  const bool issuer = (wave == 0);

  if (issuer) {
    tdm_load_tile(xb + m0 * K_PACK, lds_addr_of(&sA[0][0]), M_TOT);
    tdm_load_tile(wb + n0 * K_PACK, lds_addr_of(&sB[0][0]), N_TOT);
  }

  for (int kt = 0; kt < nKT; ++kt) {
    if (issuer) {
      if (kt + 1 < nKT) {
        const long k0n = (long)(kt + 1) * BK;
        tdm_load_tile(xb + m0 * K_PACK + k0n, lds_addr_of(&sA[(kt + 1) & 1][0]), M_TOT);
        tdm_load_tile(wb + n0 * K_PACK + k0n, lds_addr_of(&sB[(kt + 1) & 1][0]), N_TOT);
        __builtin_amdgcn_s_wait_tensorcnt(2);  // tiles kt done, kt+1 in flight
      } else {
        __builtin_amdgcn_s_wait_tensorcnt(0);
      }
    }
    __syncthreads();  // tiles kt visible to all waves
#pragma unroll
    for (int kk = 0; kk < BK; kk += 32)
      wmma_step(&sA[kt & 1][0], &sB[kt & 1][0], wm, wn, lane, kk, acc);
    __syncthreads();  // compute done before this buffer is overwritten at kt+2
  }

  const int rofs = (lane >> 4) * 8;
  const int ccol = lane & 15;
#pragma unroll
  for (int mi = 0; mi < 4; ++mi)
#pragma unroll
    for (int ni = 0; ni < 2; ++ni) {
      const long rbase = m0 + wm * 64 + mi * 16 + rofs;
      const long c     = n0 + wn * 32 + ni * 16 + ccol;
#pragma unroll
      for (int v = 0; v < 8; ++v)
        out[(rbase + v) * (long)N_TOT + c] = acc[mi][ni][v];
    }
}
#endif  // HAVE_TDM

// ---------------------------------------------------------------------------
// Fallback: fused single-pass kernel (fp32->bf16 + NF4 LUT staging in-loop).
// ---------------------------------------------------------------------------
__global__ __launch_bounds__(256, 2)
void qlora_fused_kernel(const float* __restrict__ x, const int* __restrict__ qw,
                        const float* __restrict__ wscale,
                        const float* __restrict__ loraB,
                        const unsigned short* __restrict__ tbuf,
                        float* __restrict__ out) {
  __shared__ __align__(16) unsigned short sA[BM * KP];
  __shared__ __align__(16) unsigned short sB[BN * KP];
  __shared__ unsigned short sLUT[16];

  const int tid  = threadIdx.x;
  const int lane = tid & 31;
  const int wave = tid >> 5;
  const int wm   = wave & 1;
  const int wn   = wave >> 1;
  const long m0  = (long)blockIdx.x * BM;
  const long n0  = (long)blockIdx.y * BN;

  if (tid < 16) sLUT[tid] = bf16_of(NF4_TAB[tid] * wscale[0]);

  v8f acc[4][2];
#pragma unroll
  for (int i = 0; i < 4; ++i)
#pragma unroll
    for (int j = 0; j < 2; ++j) acc[i][j] = (v8f){0, 0, 0, 0, 0, 0, 0, 0};

  const int lc = tid & 15;
  const int lr = tid >> 4;

  const int nKT = K_TOT / BK;
  for (int kt = 0; kt < nKT; ++kt) {
    const long k0 = (long)kt * BK;
#pragma unroll
    for (int it = 0; it < 8; ++it) {
      const int row = lr + it * 16;
      const float4 f = *(const float4*)(x + (m0 + row) * K_TOT + k0 + lc * 4);
      uint2 p;
      p.x = pack2_bf16(f.x, f.y);
      p.y = pack2_bf16(f.z, f.w);
      *(uint2*)(&sA[row * KP + lc * 4]) = p;
    }
#pragma unroll
    for (int it = 0; it < 8; ++it) {
      const int row = lr + it * 16;
      const int4 qi = *(const int4*)(qw + (n0 + row) * K_TOT + k0 + lc * 4);
      uint2 p;
      p.x = (uint32_t)sLUT[qi.x & 15] | ((uint32_t)sLUT[qi.y & 15] << 16);
      p.y = (uint32_t)sLUT[qi.z & 15] | ((uint32_t)sLUT[qi.w & 15] << 16);
      *(uint2*)(&sB[row * KP + lc * 4]) = p;
    }
    if (kt + 1 < nKT) {
      __builtin_prefetch(x + (m0 + lr) * K_TOT + k0 + BK + lc * 4, 0, 0);
      __builtin_prefetch(qw + (n0 + lr) * K_TOT + k0 + BK + lc * 4, 0, 0);
    }
    __syncthreads();
#pragma unroll
    for (int kk = 0; kk < BK; kk += 32)
      wmma_step(sA, sB, wm, wn, lane, kk, acc);
    __syncthreads();
  }
  {
    const int cc = tid & 7;
    const int rr = tid >> 3;
#pragma unroll
    for (int it = 0; it < 4; ++it) {
      const int row = rr + it * 32;
      const uint4 q = *(const uint4*)(tbuf + (m0 + row) * R_LORA + cc * 8);
      *(uint4*)(&sA[row * KP + cc * 8]) = q;
    }
#pragma unroll
    for (int it = 0; it < 8; ++it) {
      const int row = lr + it * 16;
      const float4 f = *(const float4*)(loraB + (n0 + row) * R_LORA + lc * 4);
      uint2 p;
      p.x = pack2_bf16(f.x, f.y);
      p.y = pack2_bf16(f.z, f.w);
      *(uint2*)(&sB[row * KP + lc * 4]) = p;
    }
    __syncthreads();
#pragma unroll
    for (int kk = 0; kk < R_LORA; kk += 32)
      wmma_step(sA, sB, wm, wn, lane, kk, acc);
  }
  const int rofs = (lane >> 4) * 8;
  const int ccol = lane & 15;
#pragma unroll
  for (int mi = 0; mi < 4; ++mi)
#pragma unroll
    for (int ni = 0; ni < 2; ++ni) {
      const long rbase = m0 + wm * 64 + mi * 16 + rofs;
      const long c     = n0 + wn * 32 + ni * 16 + ccol;
#pragma unroll
      for (int v = 0; v < 8; ++v)
        out[(rbase + v) * (long)N_TOT + c] = acc[mi][ni][v];
    }
}

extern "C" void kernel_launch(void* const* d_in, const int* in_sizes, int n_in,
                              void* d_out, int out_size, void* d_ws, size_t ws_size,
                              hipStream_t stream) {
  (void)in_sizes; (void)n_in; (void)out_size;
  const float* x      = (const float*)d_in[0];
  const int*   qw     = (const int*)d_in[1];
  const float* wscale = (const float*)d_in[2];
  const float* loraA  = (const float*)d_in[3];
  const float* loraB  = (const float*)d_in[4];
  float* out = (float*)d_out;

  const size_t xb_bytes = (size_t)M_TOT * K_PACK * 2;   // ~68 MB
  const size_t wb_bytes = (size_t)N_TOT * K_PACK * 2;   // ~34 MB
  const size_t ab_bytes = (size_t)R_LORA * K_TOT * 2;   // 512 KB

#if HAVE_TDM
  if (ws_size >= xb_bytes + wb_bytes + ab_bytes) {
    unsigned short* xb = (unsigned short*)d_ws;
    unsigned short* wb = (unsigned short*)((char*)d_ws + xb_bytes);
    unsigned short* ab = (unsigned short*)((char*)d_ws + xb_bytes + wb_bytes);
    pack_x_kernel<<<dim3((M_TOT * 512) / 256), 256, 0, stream>>>(x, xb);
    pack_w_kernel<<<dim3((N_TOT * 512) / 256), 256, 0, stream>>>(qw, wscale, wb);
    pack_wtail_kernel<<<dim3((N_TOT * 8) / 256), 256, 0, stream>>>(loraB, wb);
    pack_a_kernel<<<dim3((R_LORA * K_TOT / 8) / 256), 256, 0, stream>>>(loraA, ab);
    lora_xa_bf16_kernel<<<dim3(M_TOT / 128), 256, 0, stream>>>(xb, ab, xb + K_TOT);
    gemm_bf16_tdm_kernel<<<dim3(M_TOT / BM, N_TOT / BN), 256, 0, stream>>>(xb, wb, out);
    return;
  }
#endif
  // Fallback: fused single-pass path (needs only 1 MB of ws for t)
  unsigned short* tbuf = (unsigned short*)d_ws;
  lora_xa_f32_kernel<<<dim3(M_TOT / 128), 256, 0, stream>>>(x, loraA, tbuf, (long)R_LORA);
  qlora_fused_kernel<<<dim3(M_TOT / BM, N_TOT / BN), 256, 0, stream>>>(
      x, qw, wscale, loraB, tbuf, out);
}
